// SeqToSeqGru_25829933318443
// MI455X (gfx1250) — compile-verified, matrix operand order checked by reading, not run on hardware
//
#include <hip/hip_runtime.h>
#include <math.h>

#define NSAMP  131072
#define MT     64     // samples per workgroup tile
#define KH     224    // padded K for 200/202 (7 x 32)
#define NH     208    // padded N for 200/202 (13 x 16)
#define K1     128    // padded 100 (4 x 32)
#define N1     112    // padded 100 (7 x 16)
#define N2     64     // padded 50  (4 x 16)

// workspace byte offsets (all 16B aligned)
#define WS_WHH  0u        // 3 * 208 * 224 bf16 = 279,552 B (gate-split W_hh)
#define WS_DW1  279552u   // 112 * 224 bf16
#define WS_DW2  329728u   // 64  * 128 bf16
#define WS_EW1  346112u   // 112 * 224 bf16
#define WS_EW2  396288u   // 208 * 128 bf16
#define WS_EWL  449536u   // 208 * 224 bf16
#define WS_H0   542720u   // NSAMP * 200 bf16 = 52,428,800 B

typedef __attribute__((ext_vector_type(8)))  __bf16 v8bf;
typedef __attribute__((ext_vector_type(16))) __bf16 v16bf;
typedef __attribute__((ext_vector_type(8)))  float  v8f;
typedef __attribute__((ext_vector_type(4)))  unsigned int u32x4;
typedef __attribute__((ext_vector_type(8)))  int i32x8;
typedef __attribute__((ext_vector_type(4)))  int i32x4;

// Fragment load for 16-bit 16x32 A (and symmetric B) layout:
// lane l: row r = l&15, hi = l>>4; VGPR0-3 hold K = hi*8 .. hi*8+7 (contig),
// VGPR4-7 hold K = 16+hi*8 .. 16+hi*8+7 (contig) -> two b128 loads.
__device__ __forceinline__ v16bf load_frag(const __bf16* mat, int row0, int k0, int ld) {
  const int lane = threadIdx.x & 31;
  const int r = lane & 15, hi = lane >> 4;
  const __bf16* p = mat + (size_t)(row0 + r) * ld + k0 + hi * 8;
  v8bf lo = *(const v8bf*)p;
  v8bf hh = *(const v8bf*)(p + 16);
  v16bf f;
#pragma unroll
  for (int i = 0; i < 8; ++i) { f[i] = lo[i]; f[i + 8] = hh[i]; }
  return f;
}

__device__ __forceinline__ v8f wmma_bf16(v16bf a, v16bf b, v8f c) {
  return __builtin_amdgcn_wmma_f32_16x16x32_bf16(false, a, false, b, (short)0, c,
                                                 false, false);
}

__device__ __forceinline__ float lrelu(float v) { return v > 0.f ? v : 0.01f * v; }
__device__ __forceinline__ float sigm(float v)  { return 1.f / (1.f + __expf(-v)); }

// ---------------------------------------------------------------------------
// Prep: convert all weights to zero-padded bf16 [n][k] layouts in workspace.
// ---------------------------------------------------------------------------
__global__ __launch_bounds__(256) void prep_kernel(
    const float* __restrict__ whh, const float* __restrict__ dw1,
    const float* __restrict__ dw2, const float* __restrict__ ew1,
    const float* __restrict__ ew2, const float* __restrict__ ewl,
    char* __restrict__ wsb) {
  const int tid = blockIdx.x * blockDim.x + threadIdx.x;
  const int nth = gridDim.x * blockDim.x;

  __bf16* p = (__bf16*)(wsb + WS_WHH);   // gate-split: [g][208][224]
  for (int i = tid; i < 3 * NH * KH; i += nth) {
    int g = i / (NH * KH), r = i % (NH * KH);
    int j = r / KH, k = r % KH;
    float v = (j < 200 && k < 200) ? whh[(size_t)(g * 200 + j) * 200 + k] : 0.f;
    p[i] = (__bf16)v;
  }
  p = (__bf16*)(wsb + WS_DW1);           // [112][224] from (100,200)
  for (int i = tid; i < N1 * KH; i += nth) {
    int j = i / KH, k = i % KH;
    p[i] = (__bf16)((j < 100 && k < 200) ? dw1[j * 200 + k] : 0.f);
  }
  p = (__bf16*)(wsb + WS_DW2);           // [64][128] from (50,100)
  for (int i = tid; i < N2 * K1; i += nth) {
    int j = i / K1, k = i % K1;
    p[i] = (__bf16)((j < 50 && k < 100) ? dw2[j * 100 + k] : 0.f);
  }
  p = (__bf16*)(wsb + WS_EW1);           // [112][224] from (100,202)
  for (int i = tid; i < N1 * KH; i += nth) {
    int j = i / KH, k = i % KH;
    p[i] = (__bf16)((j < 100 && k < 202) ? ew1[j * 202 + k] : 0.f);
  }
  p = (__bf16*)(wsb + WS_EW2);           // [208][128] from (202,100)
  for (int i = tid; i < NH * K1; i += nth) {
    int j = i / K1, k = i % K1;
    p[i] = (__bf16)((j < 202 && k < 100) ? ew2[j * 100 + k] : 0.f);
  }
  p = (__bf16*)(wsb + WS_EWL);           // [208][224] from (200,202)
  for (int i = tid; i < NH * KH; i += nth) {
    int j = i / KH, k = i % KH;
    p[i] = (__bf16)((j < 200 && k < 202) ? ewl[j * 202 + k] : 0.f);
  }
}

// ---------------------------------------------------------------------------
// Encoder: DFT features + 3 WMMA GEMMs -> h0 (bf16) in workspace.
// LDS: tw(832) | sig 64x224 bf16 | y1 64x128 bf16 | y2 64x224 bf16 = 74,560 B
// ---------------------------------------------------------------------------
__global__ __launch_bounds__(256) void encoder_kernel(
    const float* __restrict__ x, const float* __restrict__ eb1,
    const float* __restrict__ eb2, const float* __restrict__ ebl,
    const __bf16* __restrict__ ew1, const __bf16* __restrict__ ew2,
    const __bf16* __restrict__ ewl, __bf16* __restrict__ h0out) {
  extern __shared__ __align__(16) char smem[];
  float2* tw  = (float2*)smem;
  __bf16* sig = (__bf16*)(smem + 832);
  __bf16* y1  = (__bf16*)(smem + 832 + MT * KH * 2);
  __bf16* y2  = (__bf16*)(smem + 832 + MT * KH * 2 + MT * K1 * 2);

  const int tid  = threadIdx.x;
  const int wave = tid >> 5;
  const int lane = tid & 31;
  const int s0   = blockIdx.x * MT;

  for (int t = tid; t < 100; t += 256) {
    float s, c;
    sincosf(-2.f * 3.14159265358979323846f * (float)t / 100.f, &s, &c);
    tw[t] = make_float2(c, s);
  }
  for (int i = tid; i < MT * KH; i += 256) { sig[i] = (__bf16)0.f; y2[i] = (__bf16)0.f; }
  for (int i = tid; i < MT * K1; i += 256) y1[i] = (__bf16)0.f;
  __syncthreads();

  // raw signal portion of sig
  for (int i = tid; i < MT * 100; i += 256) {
    int s = i / 100, t = i % 100;
    sig[s * KH + t] = (__bf16)x[(size_t)(s0 + s) * 100 + t];
  }
  // direct rDFT (norm='forward'): |F| and angle
  for (int i = tid; i < MT * 51; i += 256) {
    int s = i / 51, b = i % 51;
    const float* xs = x + (size_t)(s0 + s) * 100;
    float re = 0.f, im = 0.f;
    for (int t = 0; t < 100; ++t) {
      float2 w = tw[(b * t) % 100];
      float v = xs[t];
      re = fmaf(v, w.x, re);
      im = fmaf(v, w.y, im);
    }
    re *= 0.01f; im *= 0.01f;
    sig[s * KH + 100 + b] = (__bf16)sqrtf(re * re + im * im);
    sig[s * KH + 151 + b] = (__bf16)atan2f(im, re);
  }
  __syncthreads();

  // GEMM1: sig(64x224) x ew1^T -> y1(64x112), lrelu
  for (int job = wave; job < 4 * (N1 / 16); job += 8) {
    int mt = job & 3, nt = job >> 2;
    v8f acc = {};
#pragma unroll
    for (int kk = 0; kk < KH / 32; ++kk)
      acc = wmma_bf16(load_frag(sig, mt * 16, kk * 32, KH),
                      load_frag(ew1, nt * 16, kk * 32, KH), acc);
    int col = nt * 16 + (lane & 15), hi = lane >> 4;
    float b = (col < 100) ? eb1[col] : 0.f;
#pragma unroll
    for (int i = 0; i < 8; ++i)
      y1[(mt * 16 + i + 8 * hi) * K1 + col] = (__bf16)lrelu(acc[i] + b);
  }
  __syncthreads();

  // GEMM2: y1(64x128) x ew2^T -> y2(64x208), lrelu
  for (int job = wave; job < 4 * (NH / 16); job += 8) {
    int mt = job & 3, nt = job >> 2;
    v8f acc = {};
#pragma unroll
    for (int kk = 0; kk < K1 / 32; ++kk)
      acc = wmma_bf16(load_frag(y1, mt * 16, kk * 32, K1),
                      load_frag(ew2, nt * 16, kk * 32, K1), acc);
    int col = nt * 16 + (lane & 15), hi = lane >> 4;
    float b = (col < 202) ? eb2[col] : 0.f;
#pragma unroll
    for (int i = 0; i < 8; ++i)
      y2[(mt * 16 + i + 8 * hi) * KH + col] = (__bf16)lrelu(acc[i] + b);
  }
  __syncthreads();

  // GEMM3: y2(64x224) x ewl^T -> h0(64x200)
  for (int job = wave; job < 4 * (NH / 16); job += 8) {
    int mt = job & 3, nt = job >> 2;
    v8f acc = {};
#pragma unroll
    for (int kk = 0; kk < KH / 32; ++kk)
      acc = wmma_bf16(load_frag(y2, mt * 16, kk * 32, KH),
                      load_frag(ewl, nt * 16, kk * 32, KH), acc);
    int col = nt * 16 + (lane & 15), hi = lane >> 4;
    if (col < 200) {
      float b = ebl[col];
#pragma unroll
      for (int i = 0; i < 8; ++i)
        h0out[(size_t)(s0 + mt * 16 + i + 8 * hi) * 200 + col] = (__bf16)(acc[i] + b);
    }
  }
}

// ---------------------------------------------------------------------------
// GRU + decoder MLP, 50 steps. W_hh gate-split resident in LDS (273 KB),
// staged by the Tensor Data Mover (one descriptor, TENSORcnt-tracked).
// LDS: whh3 279,552 | hbuf 64x224 bf16 | ybuf 64x128 bf16 | din 64 f32
// ---------------------------------------------------------------------------
__global__ __launch_bounds__(256) void gru_kernel(
    const float* __restrict__ x, const float* __restrict__ wih,
    const float* __restrict__ bih, const float* __restrict__ bhh,
    const float* __restrict__ db1, const float* __restrict__ db2,
    const float* __restrict__ dw3, const float* __restrict__ db3,
    const __bf16* __restrict__ whh_g, const __bf16* __restrict__ dw1,
    const __bf16* __restrict__ dw2, const __bf16* __restrict__ h0g,
    float* __restrict__ out) {
  extern __shared__ __align__(16) char smem[];
  __bf16* whhR = (__bf16*)smem;                 // [208][224] at LDS offset 0
  __bf16* whhZ = whhR + NH * KH;
  __bf16* whhN = whhZ + NH * KH;
  __bf16* hbuf = (__bf16*)(smem + 279552);      // [64][224]
  __bf16* ybuf = (__bf16*)(smem + 308224);      // [64][128]
  float*  din  = (float*)(smem + 324608);       // [64]

  const int tid  = threadIdx.x;
  const int wave = tid >> 5;
  const int lane = tid & 31;
  const int s0   = blockIdx.x * MT;

#if __has_builtin(__builtin_amdgcn_tensor_load_to_lds)
  // --- TDM: one descriptor moves all 279,552 B of gate-split W_hh to LDS.
  // D# group0: count=1 | lds_addr=0 (only dynamic LDS) | 57b global addr | type=2
  // D# group1: data_size=3 (8B) | tensor_dim0=tile_dim0=34,944 | dim1=1 | stride
  if (wave == 0) {
    const unsigned long long ga = (unsigned long long)(uintptr_t)whh_g;
    const unsigned ne = 279552u / 8u;   // 34,944 8-byte elements
    u32x4 g0;
    g0[0] = 1u;                                            // count=1 (valid)
    g0[1] = 0u;                                            // lds_addr
    g0[2] = (unsigned)(ga & 0xFFFFFFFFu);                  // global_addr[31:0]
    g0[3] = (unsigned)((ga >> 32) & 0x01FFFFFFu) | (2u << 30); // addr[56:32]|type=2
    i32x8 g1;
    g1[0] = (int)(3u << 16);                               // data_size=8B
    g1[1] = (int)((ne & 0xFFFFu) << 16);                   // tensor_dim0[15:0]
    g1[2] = (int)(((ne >> 16) & 0xFFFFu) | (1u << 16));    // dim0[31:16]|dim1=1
    g1[3] = (int)((ne & 0xFFFFu) << 16);                   // dim1 hi=0 | tile_dim0
    g1[4] = 1;                                             // tile_dim1=1, tile_dim2=0
    g1[5] = (int)ne;                                       // tensor_dim0_stride lo
    g1[6] = 0;
    g1[7] = 0;
    i32x4 gz4 = {0, 0, 0, 0};
    i32x8 gz8 = {0, 0, 0, 0, 0, 0, 0, 0};
    __builtin_amdgcn_tensor_load_to_lds(g0, g1, gz4, gz4, gz8, 0);
  }
#else
  for (int i = tid; i < 17472; i += 256)
    ((uint4*)whhR)[i] = ((const uint4*)whh_g)[i];
#endif

  // h0 -> hbuf (zero-padded K), zero ybuf, din = last encoder input sample
  // (overlaps with the TDM transfer)
  for (int i = tid; i < MT * KH; i += 256) {
    int s = i / KH, k = i % KH;
    hbuf[i] = (k < 200) ? h0g[(size_t)(s0 + s) * 200 + k] : (__bf16)0.f;
  }
  for (int i = tid; i < MT * K1; i += 256) ybuf[i] = (__bf16)0.f;
  if (tid < MT) din[tid] = x[(size_t)(s0 + tid) * 100 + 99];

#if __has_builtin(__builtin_amdgcn_tensor_load_to_lds)
  if (wave == 0) __builtin_amdgcn_s_wait_tensorcnt(0);
#endif
  __syncthreads();

  for (int step = 0; step < 50; ++step) {
    // --- phase 1: gh GEMMs + gates, h_new kept in registers ---
    float hn[7][8];
#pragma unroll
    for (int jj = 0; jj < 7; ++jj) {
      int job = wave + jj * 8;            // 52 jobs: 4 m-tiles x 13 n-tiles
      if (job < 52) {
        int mt = job & 3, nt = job >> 2;
        v8f aR = {}, aZ = {}, aN = {};
#pragma unroll
        for (int kk = 0; kk < KH / 32; ++kk) {
          v16bf a = load_frag(hbuf, mt * 16, kk * 32, KH);
          aR = wmma_bf16(a, load_frag(whhR, nt * 16, kk * 32, KH), aR);
          aZ = wmma_bf16(a, load_frag(whhZ, nt * 16, kk * 32, KH), aZ);
          aN = wmma_bf16(a, load_frag(whhN, nt * 16, kk * 32, KH), aN);
        }
        int col = nt * 16 + (lane & 15), hi = lane >> 4;
        int j = col < 200 ? col : 199;    // clamp pad lanes (results unused)
        float wir = wih[j], wiz = wih[j + 200], win = wih[j + 400];
        float br = bih[j] + bhh[j];
        float bz = bih[j + 200] + bhh[j + 200];
        float bn = bih[j + 400], bhn = bhh[j + 400];
#pragma unroll
        for (int i = 0; i < 8; ++i) {
          int m = mt * 16 + i + 8 * hi;
          float d = din[m];
          float r = sigm(aR[i] + fmaf(d, wir, br));
          float z = sigm(aZ[i] + fmaf(d, wiz, bz));
          float n = tanhf(fmaf(d, win, bn) + r * (aN[i] + bhn));
          float hold = (float)hbuf[m * KH + col];
          hn[jj][i] = (1.f - z) * n + z * hold;
        }
      }
    }
    __syncthreads();                       // all reads of old h done
#pragma unroll
    for (int jj = 0; jj < 7; ++jj) {
      int job = wave + jj * 8;
      if (job < 52) {
        int mt = job & 3, nt = job >> 2;
        int col = nt * 16 + (lane & 15), hi = lane >> 4;
        if (col < 200)
#pragma unroll
          for (int i = 0; i < 8; ++i)
            hbuf[(mt * 16 + i + 8 * hi) * KH + col] = (__bf16)hn[jj][i];
      }
    }
    __syncthreads();

    // --- decoder layer 1: h_new(64x224) x dw1^T -> y1(64x112), lrelu ---
    for (int job = wave; job < 28; job += 8) {
      int mt = job & 3, nt = job >> 2;
      v8f acc = {};
#pragma unroll
      for (int kk = 0; kk < KH / 32; ++kk)
        acc = wmma_bf16(load_frag(hbuf, mt * 16, kk * 32, KH),
                        load_frag(dw1, nt * 16, kk * 32, KH), acc);
      int col = nt * 16 + (lane & 15), hi = lane >> 4;
      float b = (col < 100) ? db1[col] : 0.f;
#pragma unroll
      for (int i = 0; i < 8; ++i)
        ybuf[(mt * 16 + i + 8 * hi) * K1 + col] = (__bf16)lrelu(acc[i] + b);
    }
    __syncthreads();

    // --- decoder layer 2: y1(64x128) x dw2^T -> y2(64x50), lrelu ---
    float y2r[2][8];
#pragma unroll
    for (int jj = 0; jj < 2; ++jj) {
      int job = wave + jj * 8;            // 16 jobs
      int mt = job & 3, nt = job >> 2;
      v8f acc = {};
#pragma unroll
      for (int kk = 0; kk < K1 / 32; ++kk)
        acc = wmma_bf16(load_frag(ybuf, mt * 16, kk * 32, K1),
                        load_frag(dw2, nt * 16, kk * 32, K1), acc);
      int col = nt * 16 + (lane & 15);
      float b = (col < 50) ? db2[col] : 0.f;
#pragma unroll
      for (int i = 0; i < 8; ++i) y2r[jj][i] = lrelu(acc[i] + b);
    }
    __syncthreads();                       // y1 reads complete, reuse ybuf
#pragma unroll
    for (int jj = 0; jj < 2; ++jj) {
      int job = wave + jj * 8;
      int mt = job & 3, nt = job >> 2;
      int col = nt * 16 + (lane & 15), hi = lane >> 4;
#pragma unroll
      for (int i = 0; i < 8; ++i)
        ybuf[(mt * 16 + i + 8 * hi) * K1 + col] = (__bf16)y2r[jj][i];
    }
    __syncthreads();

    // --- decoder layer 3: dot(y2[0..49], dw3) + b3 -> output & next din ---
    if (tid < MT) {
      float sum = db3[0];
      for (int c = 0; c < 50; ++c)
        sum = fmaf((float)ybuf[tid * K1 + c], dw3[c], sum);
      out[(size_t)(s0 + tid) * 50 + step] = sum;
      din[tid] = sum;
    }
    __syncthreads();
  }
}

// ---------------------------------------------------------------------------
extern "C" void kernel_launch(void* const* d_in, const int* in_sizes, int n_in,
                              void* d_out, int out_size, void* d_ws, size_t ws_size,
                              hipStream_t stream) {
  const float* x    = (const float*)d_in[0];
  const float* ew1f = (const float*)d_in[1];
  const float* eb1  = (const float*)d_in[2];
  const float* ew2f = (const float*)d_in[3];
  const float* eb2  = (const float*)d_in[4];
  const float* ewlf = (const float*)d_in[5];
  const float* ebl  = (const float*)d_in[6];
  const float* wih  = (const float*)d_in[7];
  const float* whhf = (const float*)d_in[8];
  const float* bih  = (const float*)d_in[9];
  const float* bhh  = (const float*)d_in[10];
  const float* dw1f = (const float*)d_in[11];
  const float* db1  = (const float*)d_in[12];
  const float* dw2f = (const float*)d_in[13];
  const float* db2  = (const float*)d_in[14];
  const float* dw3  = (const float*)d_in[15];
  const float* db3  = (const float*)d_in[16];
  char*  wsb = (char*)d_ws;
  float* out = (float*)d_out;

  prep_kernel<<<256, 256, 0, stream>>>(whhf, dw1f, dw2f, ew1f, ew2f, ewlf, wsb);

  const size_t enc_lds = 832 + (size_t)MT * KH * 2 * 2 + (size_t)MT * K1 * 2; // 74,560
  (void)hipFuncSetAttribute((const void*)encoder_kernel,
                            hipFuncAttributeMaxDynamicSharedMemorySize, (int)enc_lds);
  encoder_kernel<<<NSAMP / MT, 256, enc_lds, stream>>>(
      x, eb1, eb2, ebl, (const __bf16*)(wsb + WS_EW1),
      (const __bf16*)(wsb + WS_EW2), (const __bf16*)(wsb + WS_EWL),
      (__bf16*)(wsb + WS_H0));

  const size_t gru_lds = 324864;
  (void)hipFuncSetAttribute((const void*)gru_kernel,
                            hipFuncAttributeMaxDynamicSharedMemorySize, (int)gru_lds);
  gru_kernel<<<NSAMP / MT, 256, gru_lds, stream>>>(
      x, wih, bih, bhh, db1, db2, dw3, db3, (const __bf16*)(wsb + WS_WHH),
      (const __bf16*)(wsb + WS_DW1), (const __bf16*)(wsb + WS_DW2),
      (const __bf16*)(wsb + WS_H0), out);
}
